// AttentionPoolingLayer_19232863552088
// MI455X (gfx1250) — compile-verified, matrix operand order checked by reading, not run on hardware
//
#include <hip/hip_runtime.h>

typedef __bf16 bf16_t;
typedef __attribute__((ext_vector_type(16))) __bf16 v16bf;
typedef __attribute__((ext_vector_type(8)))  __bf16 v8bf;
typedef __attribute__((ext_vector_type(8)))  float  v8f;

#define BS   2048
#define SEQ  200
#define DIM  64
#define DIN  256
#define H1   32
#define H2   16
#define ROWS (BS * SEQ)   // 409600

// ---------------------------------------------------------------------------
// K0: repack W1 (256x32) and W2 (32x16) into bf16 WMMA B-fragment layout.
// Fragment layout per 32-K step: lane L (<16) holds N=L, K = {0..7, 16..23};
// lane L (>=16) holds N=L-16, K = {8..15, 24..31}; 16 bf16 contiguous per lane.
// ---------------------------------------------------------------------------
__global__ __launch_bounds__(256) void pack_weights_kernel(
    const float* __restrict__ W1, const float* __restrict__ W2,
    bf16_t* __restrict__ w1f, bf16_t* __restrict__ w2f) {
  int tid = threadIdx.x;
  // W1 fragments: 8 ksteps * 2 ntiles * 32 lanes * 16 elems = 8192
  for (int idx = tid; idx < 8 * 2 * 32 * 16; idx += 256) {
    int e    = idx & 15;
    int lane = (idx >> 4) & 31;
    int n    = (idx >> 9) & 1;
    int kk   = idx >> 10;
    int koff = (lane < 16) ? 0 : 8;
    int k    = kk * 32 + ((e < 8) ? (koff + e) : (16 + koff + (e - 8)));
    int ncol = n * 16 + (lane & 15);
    w1f[idx] = (bf16_t)W1[k * H1 + ncol];
  }
  // W2 fragments: 1 kstep * 1 ntile * 32 lanes * 16 elems = 512
  for (int idx = tid; idx < 32 * 16; idx += 256) {
    int e    = idx & 15;
    int lane = idx >> 4;
    int koff = (lane < 16) ? 0 : 8;
    int k    = (e < 8) ? (koff + e) : (16 + koff + (e - 8));
    int ncol = lane & 15;
    w2f[idx] = (bf16_t)W2[k * H2 + ncol];
  }
}

// ---------------------------------------------------------------------------
// K1: X1 = concat[q, ub, q-ub, q*ub] @ W1 + b1   (409600 x 256 x 32), bf16 WMMA
// 64 rows per block, 4 waves (wave32), each wave: 16 rows, 8 K-steps, 2 N-tiles
// ---------------------------------------------------------------------------
__global__ __launch_bounds__(128) void gemm1_kernel(
    const float* __restrict__ query, const float* __restrict__ ub,
    const float* __restrict__ b1, const bf16_t* __restrict__ w1f,
    float* __restrict__ X1) {
  __shared__ bf16_t tileA[64][DIN];   // 32 KB, bf16 staged attn_input tile
  int tid     = threadIdx.x;
  int rowBase = blockIdx.x * 64;

  // Stage: build attn_input tile in LDS (each base element expands to 4 cols)
  for (int i = 0; i < 32; ++i) {
    int idx = tid + i * 128;            // 64 rows * 64 cols
    int rl  = idx >> 6;
    int c   = idx & 63;
    int r   = rowBase + rl;
    int b   = r / SEQ;
    float qv = query[b * DIM + c];
    float uv = ub[(size_t)r * DIM + c];
    tileA[rl][c]       = (bf16_t)qv;
    tileA[rl][64 + c]  = (bf16_t)uv;
    tileA[rl][128 + c] = (bf16_t)(qv - uv);
    tileA[rl][192 + c] = (bf16_t)(qv * uv);
  }
  __syncthreads();

  int lane  = tid & 31;
  int w     = tid >> 5;
  int mrow  = lane & 15;
  int koff  = (lane < 16) ? 0 : 8;
  int lrow  = w * 16 + mrow;            // local A row for this lane

  v8f c0 = {};
  v8f c1 = {};
#pragma unroll
  for (int kk = 0; kk < 8; ++kk) {
    v8bf lo = *(const v8bf*)&tileA[lrow][kk * 32 + koff];
    v8bf hi = *(const v8bf*)&tileA[lrow][kk * 32 + 16 + koff];
    v16bf a;
#pragma unroll
    for (int i = 0; i < 8; ++i) { a[i] = lo[i]; a[8 + i] = hi[i]; }
    v16bf bf0 = *(const v16bf*)&w1f[((kk * 2 + 0) * 32 + lane) * 16];
    v16bf bf1 = *(const v16bf*)&w1f[((kk * 2 + 1) * 32 + lane) * 16];
    c0 = __builtin_amdgcn_wmma_f32_16x16x32_bf16(false, a, false, bf0,
                                                 (short)0, c0, false, false);
    c1 = __builtin_amdgcn_wmma_f32_16x16x32_bf16(false, a, false, bf1,
                                                 (short)0, c1, false, false);
  }

  // D layout: VGPR j -> M = j (lanes 0-15) / j+8 (lanes 16-31); N = lane%16
  int col0   = lane & 15;
  float bia0 = b1[col0];
  float bia1 = b1[col0 + 16];
  size_t rb  = (size_t)rowBase + w * 16 + ((lane < 16) ? 0 : 8);
#pragma unroll
  for (int j = 0; j < 8; ++j) {
    X1[(rb + j) * H1 + col0]      = c0[j] + bia0;
    X1[(rb + j) * H1 + col0 + 16] = c1[j] + bia1;
  }
}

// ---------------------------------------------------------------------------
// K2: batch stats of X1 over axis 0: mu, rstd per (s, ch) — ch = 32
// ---------------------------------------------------------------------------
__global__ __launch_bounds__(256) void stats1_kernel(
    const float* __restrict__ X1, float* __restrict__ mu1,
    float* __restrict__ rstd1) {
  __shared__ float ssum[8][H1];
  __shared__ float ssq[8][H1];
  int s   = blockIdx.x;
  int tid = threadIdx.x;
  int ch  = tid & 31;
  int grp = tid >> 5;
  float sum = 0.f, sq = 0.f;
  for (int b = grp; b < BS; b += 8) {
    float v = X1[((size_t)b * SEQ + s) * H1 + ch];
    sum += v;
    sq  += v * v;
  }
  ssum[grp][ch] = sum;
  ssq[grp][ch]  = sq;
  __syncthreads();
  if (tid < H1) {
    float S = 0.f, Q = 0.f;
#pragma unroll
    for (int g = 0; g < 8; ++g) { S += ssum[g][tid]; Q += ssq[g][tid]; }
    float mu  = S * (1.0f / BS);
    float var = (Q - (float)BS * mu * mu) * (1.0f / (BS - 1));   // ddof=1
    mu1[s * H1 + tid]   = mu;
    rstd1[s * H1 + tid] = rsqrtf(fmaxf(var, 0.0f) + 1e-9f);
  }
}

// ---------------------------------------------------------------------------
// K3: X2 = dice(X1) @ W2 + b2   (409600 x 32 x 16), single-K-step bf16 WMMA
// ---------------------------------------------------------------------------
__global__ __launch_bounds__(128) void gemm2_kernel(
    const float* __restrict__ X1, const float* __restrict__ mu1,
    const float* __restrict__ rstd1, const float* __restrict__ alpha1,
    const float* __restrict__ b2, const bf16_t* __restrict__ w2f,
    float* __restrict__ X2) {
  __shared__ bf16_t tileH[64][H1];   // 4 KB
  int tid     = threadIdx.x;
  int rowBase = blockIdx.x * 64;
  float alpha = alpha1[0];

  for (int i = 0; i < 16; ++i) {
    int idx  = tid + i * 128;          // 64 rows * 32 cols
    int rl   = idx >> 5;
    int ch   = idx & 31;
    size_t r = (size_t)rowBase + rl;
    int s    = (int)(r % SEQ);
    float x  = X1[r * H1 + ch];
    float mu = mu1[s * H1 + ch];
    float rs = rstd1[s * H1 + ch];
    float p  = 1.0f / (1.0f + __expf(-(x - mu) * rs));
    float h  = alpha * x * (1.0f - p) + x * p;
    tileH[rl][ch] = (bf16_t)h;
  }
  __syncthreads();

  int lane = tid & 31;
  int w    = tid >> 5;
  int mrow = lane & 15;
  int koff = (lane < 16) ? 0 : 8;
  int lrow = w * 16 + mrow;

  v8bf lo = *(const v8bf*)&tileH[lrow][koff];
  v8bf hi = *(const v8bf*)&tileH[lrow][16 + koff];
  v16bf a;
#pragma unroll
  for (int i = 0; i < 8; ++i) { a[i] = lo[i]; a[8 + i] = hi[i]; }
  v16bf bfrag = *(const v16bf*)&w2f[lane * 16];
  v8f c = {};
  c = __builtin_amdgcn_wmma_f32_16x16x32_bf16(false, a, false, bfrag,
                                              (short)0, c, false, false);

  int col    = lane & 15;
  float bias = b2[col];
  size_t rb  = (size_t)rowBase + w * 16 + ((lane < 16) ? 0 : 8);
#pragma unroll
  for (int j = 0; j < 8; ++j) X2[(rb + j) * H2 + col] = c[j] + bias;
}

// ---------------------------------------------------------------------------
// K4: batch stats of X2 — ch = 16
// ---------------------------------------------------------------------------
__global__ __launch_bounds__(256) void stats2_kernel(
    const float* __restrict__ X2, float* __restrict__ mu2,
    float* __restrict__ rstd2) {
  __shared__ float ssum[16][H2];
  __shared__ float ssq[16][H2];
  int s   = blockIdx.x;
  int tid = threadIdx.x;
  int ch  = tid & 15;
  int grp = tid >> 4;
  float sum = 0.f, sq = 0.f;
  for (int b = grp; b < BS; b += 16) {
    float v = X2[((size_t)b * SEQ + s) * H2 + ch];
    sum += v;
    sq  += v * v;
  }
  ssum[grp][ch] = sum;
  ssq[grp][ch]  = sq;
  __syncthreads();
  if (tid < H2) {
    float S = 0.f, Q = 0.f;
#pragma unroll
    for (int g = 0; g < 16; ++g) { S += ssum[g][tid]; Q += ssq[g][tid]; }
    float mu  = S * (1.0f / BS);
    float var = (Q - (float)BS * mu * mu) * (1.0f / (BS - 1));
    mu2[s * H2 + tid]   = mu;
    rstd2[s * H2 + tid] = rsqrtf(fmaxf(var, 0.0f) + 1e-9f);
  }
}

// ---------------------------------------------------------------------------
// K5: attns = (dice(X2) @ W3 + b3) * mask ; output = sum_s ub * attns
// One block per batch element.
// ---------------------------------------------------------------------------
__global__ __launch_bounds__(256) void attn_pool_kernel(
    const float* __restrict__ ub, const float* __restrict__ mask,
    const float* __restrict__ X2, const float* __restrict__ mu2,
    const float* __restrict__ rstd2, const float* __restrict__ alpha2,
    const float* __restrict__ W3, const float* __restrict__ b3,
    float* __restrict__ out, float* __restrict__ attns) {
  __shared__ float attnS[SEQ];
  __shared__ float part[4][DIM];
  int b   = blockIdx.x;
  int tid = threadIdx.x;
  float alpha = alpha2[0];

  if (tid < SEQ) {
    int s    = tid;
    size_t r = (size_t)b * SEQ + s;
    float acc = b3[0];
#pragma unroll
    for (int ch = 0; ch < H2; ++ch) {
      float x  = X2[r * H2 + ch];
      float mu = mu2[s * H2 + ch];
      float rs = rstd2[s * H2 + ch];
      float p  = 1.0f / (1.0f + __expf(-(x - mu) * rs));
      float h  = alpha * x * (1.0f - p) + x * p;
      acc += h * W3[ch];
    }
    float a  = acc * mask[r];
    attns[r] = a;
    attnS[s] = a;
  }
  __syncthreads();

  int d   = tid & 63;
  int grp = tid >> 6;
  float acc = 0.f;
  for (int s = grp; s < SEQ; s += 4)
    acc += ub[((size_t)b * SEQ + s) * DIM + d] * attnS[s];
  part[grp][d] = acc;
  __syncthreads();
  if (tid < DIM)
    out[(size_t)b * DIM + tid] =
        part[0][tid] + part[1][tid] + part[2][tid] + part[3][tid];
}

// ---------------------------------------------------------------------------
extern "C" void kernel_launch(void* const* d_in, const int* in_sizes, int n_in,
                              void* d_out, int out_size, void* d_ws,
                              size_t ws_size, hipStream_t stream) {
  (void)in_sizes; (void)n_in; (void)out_size; (void)ws_size;
  const float* query  = (const float*)d_in[0];
  const float* ub     = (const float*)d_in[1];
  const float* mask   = (const float*)d_in[2];
  const float* W1     = (const float*)d_in[3];
  const float* b1     = (const float*)d_in[4];
  const float* alpha1 = (const float*)d_in[5];
  const float* W2     = (const float*)d_in[6];
  const float* b2     = (const float*)d_in[7];
  const float* alpha2 = (const float*)d_in[8];
  const float* W3     = (const float*)d_in[9];
  const float* b3     = (const float*)d_in[10];

  char* ws   = (char*)d_ws;
  size_t off = 0;
  auto wsalloc = [&](size_t bytes) -> void* {
    void* p = (void*)(ws + off);
    off += (bytes + 255) & ~(size_t)255;
    return p;
  };
  float*  X1    = (float*)wsalloc((size_t)ROWS * H1 * sizeof(float)); // 52.4 MB
  float*  X2    = (float*)wsalloc((size_t)ROWS * H2 * sizeof(float)); // 26.2 MB
  float*  mu1   = (float*)wsalloc((size_t)SEQ * H1 * sizeof(float));
  float*  rstd1 = (float*)wsalloc((size_t)SEQ * H1 * sizeof(float));
  float*  mu2   = (float*)wsalloc((size_t)SEQ * H2 * sizeof(float));
  float*  rstd2 = (float*)wsalloc((size_t)SEQ * H2 * sizeof(float));
  bf16_t* w1f   = (bf16_t*)wsalloc((size_t)8 * 2 * 32 * 16 * sizeof(bf16_t));
  bf16_t* w2f   = (bf16_t*)wsalloc((size_t)32 * 16 * sizeof(bf16_t));

  float* out   = (float*)d_out;
  float* attns = out + (size_t)BS * DIM;   // outputs concatenated in order

  pack_weights_kernel<<<1, 256, 0, stream>>>(W1, W2, w1f, w2f);
  gemm1_kernel<<<ROWS / 64, 128, 0, stream>>>(query, ub, b1, w1f, X1);
  stats1_kernel<<<SEQ, 256, 0, stream>>>(X1, mu1, rstd1);
  gemm2_kernel<<<ROWS / 64, 128, 0, stream>>>(X1, mu1, rstd1, alpha1, b2, w2f, X2);
  stats2_kernel<<<SEQ, 256, 0, stream>>>(X2, mu2, rstd2);
  attn_pool_kernel<<<BS, 256, 0, stream>>>(ub, mask, X2, mu2, rstd2, alpha2,
                                           W3, b3, out, attns);
}
